// Attention_33689723470500
// MI455X (gfx1250) — matrix operand order repaired and checked by reading, not compiled, not run on hardware
//
#include <hip/hip_runtime.h>

typedef __attribute__((ext_vector_type(16))) _Float16 v16h;
typedef __attribute__((ext_vector_type(8)))  _Float16 v8h;
typedef __attribute__((ext_vector_type(4)))  _Float16 v4h;
typedef __attribute__((ext_vector_type(8)))  float    v8f;
typedef __attribute__((ext_vector_type(4)))  float    v4f;

namespace {
constexpr int S_TOK = 2048;   // tokens
constexpr int H     = 32;     // query heads
constexpr int HKV   = 8;      // kv heads
constexpr int G     = 4;      // query heads per kv head
constexpr int D     = 128;    // head dim
constexpr int BM    = 32;     // q tile rows (== DIFF_BS -> mask is pure truncation)
constexpr int BN    = 64;     // kv tile (2 diffusion blocks per step)
constexpr float SCALE  = 0.08838834764831845f;   // 1/sqrt(128)
constexpr float LOG2E  = 1.4426950408889634f;
constexpr float CSCALE = SCALE * LOG2E;          // fold scale into log2 domain
// LDS strides (f16 elements): multiples of 8 keep ds_load_b128 16B-aligned,
// odd multiples of 8 (mod 64 DWORD banks) spread banks.
constexpr int KST = 136;      // sK row stride  [key][d]
constexpr int VTS = 72;       // sVt row stride [d][key]
constexpr int PST = 72;       // sP row stride  [wave][row][key]
}

// DPP16 lane reductions within each 16-lane half (VALU pipe, no DS traffic).
// dpp_ctrl must be an immediate -> template parameter.
template <int CTRL>
__device__ __forceinline__ float dpp_mov(float x) {
  int r = __builtin_amdgcn_update_dpp(0, __builtin_bit_cast(int, x),
                                      CTRL, 0xf, 0xf, true);
  return __builtin_bit_cast(float, r);
}
__device__ __forceinline__ float rowmax16(float x) {
  x = fmaxf(x, dpp_mov<0x0B1>(x));  // quad_perm [1,0,3,2]  (xor 1)
  x = fmaxf(x, dpp_mov<0x04E>(x));  // quad_perm [2,3,0,1]  (xor 2)
  x = fmaxf(x, dpp_mov<0x141>(x));  // row_half_mirror      (merge quads of 8)
  x = fmaxf(x, dpp_mov<0x140>(x));  // row_mirror           (merge halves of 16)
  return x;
}
__device__ __forceinline__ float rowsum16(float x) {
  x += dpp_mov<0x0B1>(x);
  x += dpp_mov<0x04E>(x);
  x += dpp_mov<0x141>(x);
  x += dpp_mov<0x140>(x);
  return x;
}

__global__ __launch_bounds__(256) void fa_gqa_kernel(
    const float* __restrict__ q, const float* __restrict__ k,
    const float* __restrict__ v, float* __restrict__ out)
{
  __shared__ _Float16 sK [BN * KST];       // K tile, row-major [key][d]
  __shared__ _Float16 sVt[D  * VTS];       // V tile, transposed [d][key]
  __shared__ _Float16 sP [8 * 16 * PST];   // per-wave P staging [wave][row][key]

  const int qblk = (int)gridDim.x - 1 - (int)blockIdx.x;  // big tiles first
  const int h    = blockIdx.y;             // kv head
  const int tid  = threadIdx.x;
  const int wave = tid >> 5;
  const int lane = tid & 31;
  const int lo16 = lane & 15;
  const int hi   = lane >> 4;              // 0 or 1 (half-wave)
  const int g    = wave >> 1;              // query head within GQA group
  const int sub  = wave & 1;               // 16-row sub-tile
  const int qhead = h * G + g;
  const int a0 = hi ? 8  : 0;              // A-operand K packing base (ISA A layout)
  const int b0 = hi ? 8  : 0;              // B-operand K packing base (ISA B layout)

  // ---- Q sub-tile: f32 -> f16 directly into WMMA A-layout registers ----
  v16h Aq[4];
  {
    const int row = qblk * BM + sub * 16 + lo16;
    const float* qp = q + (size_t)row * (H * D) + (size_t)qhead * D;
#pragma unroll
    for (int c = 0; c < 4; ++c) {
      v4f x0 = *(const v4f*)(qp + 32 * c + a0);
      v4f x1 = *(const v4f*)(qp + 32 * c + a0 + 4);
      v4f x2 = *(const v4f*)(qp + 32 * c + a0 + 16);
      v4f x3 = *(const v4f*)(qp + 32 * c + a0 + 20);
      v16h a;
#pragma unroll
      for (int e = 0; e < 4; ++e) {
        a[e]      = (_Float16)x0[e];
        a[4 + e]  = (_Float16)x1[e];
        a[8 + e]  = (_Float16)x2[e];
        a[12 + e] = (_Float16)x3[e];
      }
      Aq[c] = a;
    }
  }

  v8f vzero = {};
  v8f O[8];
#pragma unroll
  for (int t = 0; t < 8; ++t) O[t] = vzero;
  float m_r[8], l_r[8];
#pragma unroll
  for (int r = 0; r < 8; ++r) { m_r[r] = -3.0e38f; l_r[r] = 0.0f; }

  _Float16* pw = &sP[wave * 16 * PST];

  const int kvlen  = (qblk + 1) * 32;      // block-causal mask == truncation
  const int nsteps = (qblk + 2) >> 1;      // ceil(kvlen / 64)

  for (int step = 0; step < nsteps; ++step) {
    const int kbase = step * BN;

    // ---- cooperative staging: K (row-major) + V (transposed), f32 -> f16 ----
    __syncthreads();   // previous iteration's readers done
#pragma unroll
    for (int j = 0; j < 8; ++j) {
      const int key = wave + 8 * j;          // 0..63, uniform per wave
      const int d4  = lane * 4;              // coalesced 512B/row per wave
      v4f kx = {}, vx = {};
      if (kbase + key < kvlen) {             // zero-fill tail (no stale NaNs)
        const size_t goff = (size_t)(kbase + key) * (HKV * D) + (size_t)h * D + d4;
        kx = *(const v4f*)(k + goff);
        vx = *(const v4f*)(v + goff);
      }
      v4h kh = { (_Float16)kx[0], (_Float16)kx[1], (_Float16)kx[2], (_Float16)kx[3] };
      *(v4h*)&sK[key * KST + d4] = kh;       // ds_store_b64, conflict-free
#pragma unroll
      for (int e = 0; e < 4; ++e)
        sVt[(d4 + e) * VTS + key] = (_Float16)vx[e];
    }
    __syncthreads();

    // ---- scores = Q * K^T : four 16x16 key tiles, K-dim (=D) chunked by 32 ----
    v8f S[4];
#pragma unroll
    for (int i = 0; i < 4; ++i) S[i] = vzero;
#pragma unroll
    for (int c = 0; c < 4; ++c) {
#pragma unroll
      for (int i = 0; i < 4; ++i) {
        v8h bl = *(const v8h*)&sK[(i * 16 + lo16) * KST + 32 * c + b0];
        v8h bh = *(const v8h*)&sK[(i * 16 + lo16) * KST + 32 * c + b0 + 16];
        v16h B = __builtin_shufflevector(bl, bh, 0,1,2,3,4,5,6,7,8,9,10,11,12,13,14,15);
        S[i] = __builtin_amdgcn_wmma_f32_16x16x32_f16(false, Aq[c], false, B,
                                                      (short)0, S[i], false, false);
      }
    }

    // tiles 2,3 (keys kbase+32..63) invalid only on the odd tail step
    const bool tail = (kbase + 32) >= kvlen;

    // ---- online softmax in WMMA C-layout (row r+8*hi, col lo16 per VGPR r) ----
#pragma unroll
    for (int r = 0; r < 8; ++r) {
      float s0 = S[0][r] * CSCALE;
      float s1 = S[1][r] * CSCALE;
      float s2 = tail ? -3.0e38f : S[2][r] * CSCALE;
      float s3 = tail ? -3.0e38f : S[3][r] * CSCALE;
      float mx = rowmax16(fmaxf(fmaxf(s0, s1), fmaxf(s2, s3)));
      const float mnew  = fmaxf(m_r[r], mx);
      const float alpha = exp2f(m_r[r] - mnew);
      const float p0 = exp2f(s0 - mnew);
      const float p1 = exp2f(s1 - mnew);
      const float p2 = exp2f(s2 - mnew);
      const float p3 = exp2f(s3 - mnew);
      const float rs = rowsum16((p0 + p1) + (p2 + p3));
      l_r[r] = l_r[r] * alpha + rs;
      m_r[r] = mnew;
#pragma unroll
      for (int t = 0; t < 8; ++t) O[t][r] *= alpha;   // O shares C-layout rows
      const int prow = r + 8 * hi;
      pw[prow * PST + lo16]      = (_Float16)p0;      // transpose P via LDS
      pw[prow * PST + 16 + lo16] = (_Float16)p1;
      pw[prow * PST + 32 + lo16] = (_Float16)p2;
      pw[prow * PST + 48 + lo16] = (_Float16)p3;
    }

    // ---- reload P in A-layout (wave-private LDS; DS ops are in-order) ----
    v8h pl0 = *(const v8h*)&pw[lo16 * PST + a0];
    v8h ph0 = *(const v8h*)&pw[lo16 * PST + a0 + 16];
    v16h Ap0 = __builtin_shufflevector(pl0, ph0, 0,1,2,3,4,5,6,7,8,9,10,11,12,13,14,15);
    v8h pl1 = *(const v8h*)&pw[lo16 * PST + 32 + a0];
    v8h ph1 = *(const v8h*)&pw[lo16 * PST + 32 + a0 + 16];
    v16h Ap1 = __builtin_shufflevector(pl1, ph1, 0,1,2,3,4,5,6,7,8,9,10,11,12,13,14,15);

    // ---- O += P * V : 8 column tiles of 16 over D=128, K=64 in two WMMAs ----
#pragma unroll
    for (int t = 0; t < 8; ++t) {
      const int dcol = 16 * t + lo16;
      v8h vl0 = *(const v8h*)&sVt[dcol * VTS + b0];
      v8h vh0 = *(const v8h*)&sVt[dcol * VTS + b0 + 16];
      v16h Bv0 = __builtin_shufflevector(vl0, vh0, 0,1,2,3,4,5,6,7,8,9,10,11,12,13,14,15);
      O[t] = __builtin_amdgcn_wmma_f32_16x16x32_f16(false, Ap0, false, Bv0,
                                                    (short)0, O[t], false, false);
      v8h vl1 = *(const v8h*)&sVt[dcol * VTS + 32 + b0];
      v8h vh1 = *(const v8h*)&sVt[dcol * VTS + 32 + b0 + 16];
      v16h Bv1 = __builtin_shufflevector(vl1, vh1, 0,1,2,3,4,5,6,7,8,9,10,11,12,13,14,15);
      O[t] = __builtin_amdgcn_wmma_f32_16x16x32_f16(false, Ap1, false, Bv1,
                                                    (short)0, O[t], false, false);
    }
  }

  // ---- epilogue: normalize and store (coalesced 64B per half-wave row) ----
#pragma unroll
  for (int r = 0; r < 8; ++r) {
    const float inv = 1.0f / l_r[r];
    const int row = qblk * BM + sub * 16 + r + 8 * hi;
    float* orow = out + (size_t)row * (H * D) + (size_t)qhead * D + lo16;
#pragma unroll
    for (int t = 0; t < 8; ++t) orow[16 * t] = O[t][r] * inv;
  }
}

extern "C" void kernel_launch(void* const* d_in, const int* in_sizes, int n_in,
                              void* d_out, int out_size, void* d_ws, size_t ws_size,
                              hipStream_t stream) {
  const float* q = (const float*)d_in[0];
  const float* k = (const float*)d_in[1];
  const float* v = (const float*)d_in[2];
  // d_in[3] (block_mask) is ignored: DIFF_BS==BM makes the block-causal mask
  // an exact truncation of the KV loop per 32-row query tile.
  float* out = (float*)d_out;
  dim3 grid(S_TOK / BM, HKV);   // 64 q-tiles x 8 kv heads
  fa_gqa_kernel<<<grid, 256, 0, stream>>>(q, k, v, out);
  (void)in_sizes; (void)n_in; (void)out_size; (void)d_ws; (void)ws_size;
}
